// LPViTBlock_26448408609183
// MI455X (gfx1250) — matrix operand order repaired and checked by reading, not compiled, not run on hardware
//
#include <hip/hip_runtime.h>

#define DEVI __device__ __forceinline__

typedef float  v8f   __attribute__((ext_vector_type(8)));
typedef __bf16 v16bf __attribute__((ext_vector_type(16)));
typedef __bf16 v8bf  __attribute__((ext_vector_type(8)));

DEVI v8f wmma_bf16(v16bf a, v16bf b, v8f c) {
  // D = A(16x32 bf16) x B(32x16 bf16) + C(16x16 f32)
  return __builtin_amdgcn_wmma_f32_16x16x32_bf16(false, a, false, b, (short)0, c, false, false);
}

DEVI float waveSum(float v) {
#pragma unroll
  for (int o = 16; o; o >>= 1) v += __shfl_xor(v, o, 32);
  return v;
}
DEVI float waveMax(float v) {
#pragma unroll
  for (int o = 16; o; o >>= 1) v = fmaxf(v, __shfl_xor(v, o, 32));
  return v;
}
// 256-thread block reduce; result broadcast to all threads. tmp = 8 floats in LDS.
DEVI float blockReduce256(float v, float* tmp, bool ismax) {
  int lane = threadIdx.x & 31, w = threadIdx.x >> 5;
#pragma unroll
  for (int o = 16; o; o >>= 1) {
    float t = __shfl_xor(v, o, 32);
    v = ismax ? fmaxf(v, t) : (v + t);
  }
  __syncthreads();
  if (lane == 0) tmp[w] = v;
  __syncthreads();
  float r = tmp[0];
#pragma unroll
  for (int i = 1; i < 8; ++i) r = ismax ? fmaxf(r, tmp[i]) : (r + tmp[i]);
  return r;
}

// ---------------- stats: per-batch mean / invstd over C*P*N = 4M elements ---------
__global__ __launch_bounds__(256) void k_stats_part(const float* __restrict__ x,
                                                    float* __restrict__ part) {
  __shared__ float tmp[8];
  size_t base = (size_t)blockIdx.x * 16384;
  float s = 0.f, q = 0.f;
#pragma unroll 4
  for (int i = 0; i < 64; ++i) {
    float v = x[base + (size_t)i * 256 + threadIdx.x];
    s += v; q += v * v;
  }
  s = blockReduce256(s, tmp, false);
  q = blockReduce256(q, tmp, false);
  if (threadIdx.x == 0) { part[(size_t)blockIdx.x * 2] = s; part[(size_t)blockIdx.x * 2 + 1] = q; }
}

__global__ __launch_bounds__(256) void k_stats_final(const float* __restrict__ part,
                                                     float* __restrict__ mv) {
  __shared__ float tmp[8];
  int b = blockIdx.x;
  float s = part[(size_t)(b * 256 + threadIdx.x) * 2];
  float q = part[(size_t)(b * 256 + threadIdx.x) * 2 + 1];
  s = blockReduce256(s, tmp, false);
  q = blockReduce256(q, tmp, false);
  if (threadIdx.x == 0) {
    const float inv = 1.0f / 4194304.0f;
    float mean = s * inv;
    float var = q * inv - mean * mean;
    mv[b * 2] = mean;
    mv[b * 2 + 1] = rsqrtf(var + 1e-5f);
  }
}

// ---------------- LN affine folding: a = g*invstd, d = beta - mean*a ----------------
__global__ void k_prep_affine(const float* __restrict__ mv,
                              const float* __restrict__ g1, const float* __restrict__ be1,
                              const float* __restrict__ g2, const float* __restrict__ be2,
                              const float* __restrict__ Wqkv1, const float* __restrict__ Wqkv2,
                              float* a1, float* d1, float* a2, float* d2,
                              float* wq1e, float* wq2e) {
  int b = blockIdx.x, c = threadIdx.x;
  float mean = mv[b * 2], is = mv[b * 2 + 1];
  float A1 = g1[c] * is; a1[b * 256 + c] = A1; d1[b * 256 + c] = be1[c] - mean * A1;
  float A2 = g2[c] * is; a2[b * 256 + c] = A2; d2[b * 256 + c] = be2[c] - mean * A2;
  wq1e[b * 256 + c] = Wqkv1[c] * A1;   // row 0 of Wqkv = q weights
  wq2e[b * 256 + c] = Wqkv2[c] * A2;
}

__global__ void k_prep_affineF(const float* __restrict__ mvF,
                               const float* __restrict__ gf, const float* __restrict__ bf,
                               float* af, float* df) {
  int b = blockIdx.x, c = threadIdx.x;
  float mean = mvF[b * 2], is = mvF[b * 2 + 1];
  float A = gf[c] * is; af[b * 256 + c] = A; df[b * 256 + c] = bf[c] - mean * A;
}

// effective bias: out[b][o] = bias0[row_off+o] + sum_k W[row_off+o][k] * d[b][k]
__global__ __launch_bounds__(256) void k_bias_eff(const float* __restrict__ Wsrc, int K, int row_off,
                                                  const float* __restrict__ bias0,
                                                  const float* __restrict__ dvec,
                                                  float* __restrict__ out) {
  __shared__ float tmp[8];
  int o = blockIdx.x, b = blockIdx.y, M = gridDim.x;
  float acc = 0.f;
  for (int k = threadIdx.x; k < K; k += 256)
    acc += Wsrc[(size_t)(row_off + o) * K + k] * dvec[(size_t)b * K + k];
  acc = blockReduce256(acc, tmp, false);
  if (threadIdx.x == 0) out[(size_t)b * M + o] = bias0[row_off + o] + acc;
}

// ---------------- pre-fragment weights into WMMA A-operand layout (bf16) -----------
// A 16x32 bf16 k order used by this code (B reads match it): lane l (m=l&15, h=l>>4), element e (0..15) -> k = 16h + e
__global__ __launch_bounds__(256) void k_build_frags(const float* __restrict__ Wsrc, int M, int K,
                                                     const float* __restrict__ colscale, // [nb][K] or null
                                                     __bf16* __restrict__ dst) {
  size_t gid = (size_t)blockIdx.x * 256 + threadIdx.x;
  size_t MK = (size_t)M * K;
  int bsel = (int)(gid / MK);
  size_t e = gid % MK;
  int fragIdx = (int)(e >> 9);
  int lane = (int)((e >> 4) & 31);
  int slot = (int)(e & 15);
  int j = slot >> 1, sbit = slot & 1;
  int MO = M >> 4;
  int ko = fragIdx / MO, mo = fragIdx % MO;
  int m = mo * 16 + (lane & 15);
  int hh = lane >> 4;
  int kl = hh * 16 + slot;
  int k = ko * 32 + kl;
  float v = Wsrc[(size_t)m * K + k];
  if (colscale) v *= colscale[(size_t)bsel * K + k];
  dst[gid] = (__bf16)v;
}

// ---------------- q channels (GEMV over C at each position) ------------------------
__global__ __launch_bounds__(256) void k_q(const float* __restrict__ x,
                                           const float* __restrict__ wq1e, const float* __restrict__ wq2e,
                                           const float* __restrict__ cq1, const float* __restrict__ cq2,
                                           float* __restrict__ q1, float* __restrict__ q2) {
  int row = blockIdx.x;          // b*64 + p
  int b = row >> 6, p = row & 63;
  int n = threadIdx.x;
  float acc1 = cq1[b], acc2 = cq2[b];
  for (int c = 0; c < 256; ++c) {
    float xv = x[((size_t)(b * 256 + c) * 64 + p) * 256 + n];
    acc1 += xv * wq1e[b * 256 + c];
    acc2 += xv * wq2e[b * 256 + c];
  }
  q1[(size_t)row * 256 + n] = acc1;
  q2[(size_t)row * 256 + n] = acc2;
}

// softmax over N per (b,p)
__global__ __launch_bounds__(256) void k_softmax2(const float* __restrict__ qv, float* __restrict__ s) {
  __shared__ float tmp[8];
  size_t row = blockIdx.x;
  float v = qv[row * 256 + threadIdx.x];
  float m = blockReduce256(v, tmp, true);
  float e = __expf(v - m);
  float S = blockReduce256(e, tmp, false);
  s[row * 256 + threadIdx.x] = e / S;
}

// softmax over P per (b,n): reads q1[b][p][n] strided, writes s1[b][n][p] contiguous
__global__ __launch_bounds__(64) void k_softmax1(const float* __restrict__ q1, float* __restrict__ s1) {
  __shared__ float tmp[2];
  int bn = blockIdx.x;           // b*256 + n
  int b = bn >> 8, n = bn & 255;
  int p = threadIdx.x;
  float v = q1[((size_t)(b * 64) + p) * 256 + n];
  float m = waveMax(v);
  __syncthreads();
  if ((threadIdx.x & 31) == 0) tmp[threadIdx.x >> 5] = m;
  __syncthreads();
  m = fmaxf(tmp[0], tmp[1]);
  float e = __expf(v - m);
  float S = waveSum(e);
  __syncthreads();
  if ((threadIdx.x & 31) == 0) tmp[threadIdx.x >> 5] = S;
  __syncthreads();
  S = tmp[0] + tmp[1];
  s1[(size_t)bn * 64 + p] = e / S;
}

// fused score-weighted reductions: one (b,c) plane per block, both branches in one x read
// t2[b][c][p] = a2*(sum_n x*s2[b,p,n]) + d2 ; t1[b][c][n] = a1*(sum_p x*s1[b,n,p]) + d1
__global__ __launch_bounds__(256) void k_traw(const float* __restrict__ x,
                                              const float* __restrict__ s1, const float* __restrict__ s2,
                                              const float* __restrict__ a1, const float* __restrict__ d1,
                                              const float* __restrict__ a2, const float* __restrict__ d2,
                                              float* __restrict__ t1, float* __restrict__ t2) {
  __shared__ float tmp[8];
  int bc = blockIdx.x;           // b*256 + c
  int b = bc >> 8;
  int tid = threadIdx.x;         // n
  size_t base = (size_t)bc * 64 * 256;
  float acc1 = 0.f, myT2 = 0.f;
  for (int p = 0; p < 64; ++p) {
    float xv = x[base + (size_t)p * 256 + tid];
    acc1 += xv * s1[((size_t)(b * 256) + tid) * 64 + p];
    float rs = xv * s2[((size_t)(b * 64) + p) * 256 + tid];
    rs = blockReduce256(rs, tmp, false);
    if (tid == p) myT2 = rs;
  }
  float A1 = a1[bc], D1 = d1[bc];
  t1[(size_t)bc * 256 + tid] = A1 * acc1 + D1;
  if (tid < 64) {
    float A2 = a2[bc], D2 = d2[bc];
    t2[(size_t)bc * 64 + tid] = A2 * myT2 + D2;
  }
}

// ctx2[b][p][c] = Wk2[c,:] @ t2[b,:,p] + bk2[c]
__global__ __launch_bounds__(64) void k_ctx2(const float* __restrict__ Wqkv2, const float* __restrict__ bqkv2,
                                             const float* __restrict__ t2, float* __restrict__ ctx2) {
  int bc = blockIdx.x; int b = bc >> 8, c = bc & 255;
  int pp = threadIdx.x;
  const float* wr = Wqkv2 + (size_t)(1 + c) * 256;
  float acc = bqkv2[1 + c];
  for (int cp = 0; cp < 256; ++cp)
    acc += wr[cp] * t2[((size_t)(b * 256) + cp) * 64 + pp];
  ctx2[((size_t)(b * 64) + pp) * 256 + c] = acc;
}

// ctx1[b][n][c] = Wk1[c,:] @ t1[b,:,n] + bk1[c]
__global__ __launch_bounds__(256) void k_ctx1(const float* __restrict__ Wqkv1, const float* __restrict__ bqkv1,
                                              const float* __restrict__ t1, float* __restrict__ ctx1) {
  int bc = blockIdx.x; int b = bc >> 8, c = bc & 255;
  int nn = threadIdx.x;
  const float* wr = Wqkv1 + (size_t)(1 + c) * 256;
  float acc = bqkv1[1 + c];
  for (int cp = 0; cp < 256; ++cp)
    acc += wr[cp] * t1[((size_t)(b * 256) + cp) * 256 + nn];
  ctx1[((size_t)(b * 256) + nn) * 256 + c] = acc;
}

// ---------------- main fused attention kernel: 64 positions x 256 channels per WG ---
// x3 = x + Wout2@(relu(v2)*ctx2) + Wout1@(relu(v1)*ctx1); also LN-stats partials of x3
__global__ __launch_bounds__(256) void k_main(const float* __restrict__ x,
                                              const __bf16* __restrict__ wv1b, const __bf16* __restrict__ wv2b,
                                              const __bf16* __restrict__ wout1f, const __bf16* __restrict__ wout2f,
                                              const float* __restrict__ bv1e, const float* __restrict__ bv2e,
                                              const float* __restrict__ bout1, const float* __restrict__ bout2,
                                              const float* __restrict__ ctx1, const float* __restrict__ ctx2,
                                              float* __restrict__ x3, float* __restrict__ partF) {
  __shared__ __attribute__((aligned(32))) __bf16 xb[64 * 272];   // [pos][chan], padded row
  __shared__ __attribute__((aligned(32))) __bf16 act[64 * 272];
  __shared__ float tmp[8];

  const int tid = threadIdx.x, lane = tid & 31, w = tid >> 5, h = lane >> 4, nl = lane & 15;
  const int T = blockIdx.x, n0 = (T & 3) * 64, row = T >> 2, p = row & 63, b = row >> 6;

  // stage x tile -> bf16 LDS [pos][chan]
#pragma unroll 4
  for (int it = 0; it < 64; ++it) {
    int c = it * 4 + (tid >> 6);
    int j = tid & 63;
    float v = x[((size_t)(b * 256 + c) * 64 + p) * 256 + n0 + j];
    xb[j * 272 + c] = (__bf16)v;
  }
  __syncthreads();

  v8f yacc[8];
#pragma unroll
  for (int ci = 0; ci < 2; ++ci)
#pragma unroll
    for (int po = 0; po < 4; ++po) {
      v8f a;
#pragma unroll
      for (int r = 0; r < 8; ++r) {
        int c = (w * 2 + ci) * 16 + 8 * h + r;
        a[r] = bout1[c] + bout2[c];
      }
      yacc[ci * 4 + po] = a;
    }

#pragma unroll 1
  for (int br = 0; br < 2; ++br) {
    const __bf16* wv  = (br == 0) ? (wv2b + (size_t)b * 65536) : (wv1b + (size_t)b * 65536);
    const __bf16* wo  = (br == 0) ? wout2f : wout1f;
    const float*  bve = (br == 0) ? bv2e : bv1e;

    if (br == 1) __syncthreads();   // previous y-GEMM reads of act complete

    // v-GEMM (LN folded into wv) -> relu * ctx -> act (bf16)
#pragma unroll
    for (int ci = 0; ci < 2; ++ci) {
      int co = w * 2 + ci;
#pragma unroll
      for (int po = 0; po < 4; ++po) {
        v8f acc;
#pragma unroll
        for (int r = 0; r < 8; ++r) acc[r] = bve[b * 256 + co * 16 + 8 * h + r];
#pragma unroll
        for (int ko = 0; ko < 8; ++ko) {
          v16bf A  = *(const v16bf*)(wv + (size_t)((ko * 16 + co) * 32 + lane) * 16);
          v16bf Bm = *(const v16bf*)(&xb[(po * 16 + nl) * 272 + ko * 32 + h * 16]);
          acc = wmma_bf16(A, Bm, acc);
        }
        v8bf pk;
#pragma unroll
        for (int r = 0; r < 8; ++r) {
          int c = co * 16 + 8 * h + r;
          float cv = (br == 0) ? ctx2[((size_t)(b * 64 + p)) * 256 + c]
                               : ctx1[((size_t)(b * 256 + (n0 + po * 16 + nl))) * 256 + c];
          float v = acc[r];
          v = v > 0.f ? v : 0.f;
          pk[r] = (__bf16)(v * cv);
        }
        *(v8bf*)(&act[(po * 16 + nl) * 272 + co * 16 + 8 * h]) = pk;
      }
    }
    __syncthreads();   // act ready

    // out-GEMM accumulate into yacc
#pragma unroll
    for (int ci = 0; ci < 2; ++ci) {
      int co = w * 2 + ci;
#pragma unroll
      for (int po = 0; po < 4; ++po) {
        v8f acc = yacc[ci * 4 + po];
#pragma unroll
        for (int ko = 0; ko < 8; ++ko) {
          v16bf A  = *(const v16bf*)(wo + (size_t)((ko * 16 + co) * 32 + lane) * 16);
          v16bf Bm = *(const v16bf*)(&act[(po * 16 + nl) * 272 + ko * 32 + h * 16]);
          acc = wmma_bf16(A, Bm, acc);
        }
        yacc[ci * 4 + po] = acc;
      }
    }
  }

  // epilogue: residual add, store x3, accumulate LN-stat partials
  float s = 0.f, q = 0.f;
#pragma unroll
  for (int ci = 0; ci < 2; ++ci) {
    int co = w * 2 + ci;
#pragma unroll
    for (int po = 0; po < 4; ++po) {
      v8f a = yacc[ci * 4 + po];
      int n = n0 + po * 16 + nl;
#pragma unroll
      for (int r = 0; r < 8; ++r) {
        int c = co * 16 + 8 * h + r;
        size_t idx = ((size_t)(b * 256 + c) * 64 + p) * 256 + n;
        float v = a[r] + x[idx];
        x3[idx] = v;
        s += v; q += v * v;
      }
    }
  }
  s = blockReduce256(s, tmp, false);
  q = blockReduce256(q, tmp, false);
  if (tid == 0) { partF[(size_t)T * 2] = s; partF[(size_t)T * 2 + 1] = q; }
}

// ---------------- FFN: x += Wf2 @ clip(Wf1b@x3 + bf1e, -1, 1) + bf2 (in place) ------
__global__ __launch_bounds__(256) void k_ffn(float* __restrict__ x3,
                                             const __bf16* __restrict__ wf1b, const __bf16* __restrict__ wf2f,
                                             const float* __restrict__ bf1e, const float* __restrict__ bf2) {
  __shared__ __attribute__((aligned(32))) __bf16 xb[32 * 272];
  __shared__ __attribute__((aligned(32))) __bf16 hb[32 * 528];

  const int tid = threadIdx.x, lane = tid & 31, w = tid >> 5, h = lane >> 4, nl = lane & 15;
  const int T = blockIdx.x, n0 = (T & 7) * 32, row = T >> 3, p = row & 63, b = row >> 6;

#pragma unroll 4
  for (int it = 0; it < 32; ++it) {
    int e = it * 256 + tid;
    int c = e >> 5, j = e & 31;
    float v = x3[((size_t)(b * 256 + c) * 64 + p) * 256 + n0 + j];
    xb[j * 272 + c] = (__bf16)v;
  }
  __syncthreads();

  const __bf16* wf1 = wf1b + (size_t)b * 131072;
  // h-GEMM: 512 out channels, clip to [-1,1]
#pragma unroll
  for (int ci = 0; ci < 4; ++ci) {
    int co = w * 4 + ci;
#pragma unroll
    for (int po = 0; po < 2; ++po) {
      v8f acc;
#pragma unroll
      for (int r = 0; r < 8; ++r) acc[r] = bf1e[b * 512 + co * 16 + 8 * h + r];
#pragma unroll
      for (int ko = 0; ko < 8; ++ko) {
        v16bf A  = *(const v16bf*)(wf1 + (size_t)((ko * 32 + co) * 32 + lane) * 16);
        v16bf Bm = *(const v16bf*)(&xb[(po * 16 + nl) * 272 + ko * 32 + h * 16]);
        acc = wmma_bf16(A, Bm, acc);
      }
      v8bf pk;
#pragma unroll
      for (int r = 0; r < 8; ++r) {
        float v = acc[r];
        v = v < -1.f ? -1.f : (v > 1.f ? 1.f : v);
        pk[r] = (__bf16)v;
      }
      *(v8bf*)(&hb[(po * 16 + nl) * 528 + co * 16 + 8 * h]) = pk;
    }
  }
  __syncthreads();

  // out-GEMM K=512
  v8f oacc[4];
#pragma unroll
  for (int ci = 0; ci < 2; ++ci) {
    int co = w * 2 + ci;
#pragma unroll
    for (int po = 0; po < 2; ++po) {
      v8f acc;
#pragma unroll
      for (int r = 0; r < 8; ++r) acc[r] = bf2[co * 16 + 8 * h + r];
#pragma unroll
      for (int ko = 0; ko < 16; ++ko) {
        v16bf A  = *(const v16bf*)(wf2f + (size_t)((ko * 16 + co) * 32 + lane) * 16);
        v16bf Bm = *(const v16bf*)(&hb[(po * 16 + nl) * 528 + ko * 32 + h * 16]);
        acc = wmma_bf16(A, Bm, acc);
      }
      oacc[ci * 2 + po] = acc;
    }
  }

#pragma unroll
  for (int ci = 0; ci < 2; ++ci) {
    int co = w * 2 + ci;
#pragma unroll
    for (int po = 0; po < 2; ++po) {
      v8f a = oacc[ci * 2 + po];
      int n = n0 + po * 16 + nl;
#pragma unroll
      for (int r = 0; r < 8; ++r) {
        int c = co * 16 + 8 * h + r;
        size_t idx = ((size_t)(b * 256 + c) * 64 + p) * 256 + n;
        x3[idx] = x3[idx] + a[r];
      }
    }
  }
}

// =====================================================================================
extern "C" void kernel_launch(void* const* d_in, const int* in_sizes, int n_in,
                              void* d_out, int out_size, void* d_ws, size_t ws_size,
                              hipStream_t stream) {
  (void)in_sizes; (void)n_in; (void)out_size; (void)ws_size;
  const float* x     = (const float*)d_in[0];
  const float* g1    = (const float*)d_in[1];
  const float* b1    = (const float*)d_in[2];
  const float* Wqkv1 = (const float*)d_in[3];
  const float* bqkv1 = (const float*)d_in[4];
  const float* Wout1 = (const float*)d_in[5];
  const float* bout1 = (const float*)d_in[6];
  const float* g2    = (const float*)d_in[7];
  const float* b2    = (const float*)d_in[8];
  const float* Wqkv2 = (const float*)d_in[9];
  const float* bqkv2 = (const float*)d_in[10];
  const float* Wout2 = (const float*)d_in[11];
  const float* bout2 = (const float*)d_in[12];
  const float* gf    = (const float*)d_in[13];
  const float* bfv   = (const float*)d_in[14];
  const float* Wf1   = (const float*)d_in[15];
  const float* bf1   = (const float*)d_in[16];
  const float* Wf2   = (const float*)d_in[17];
  const float* bf2   = (const float*)d_in[18];
  float* xout = (float*)d_out;

  char* wsb = (char*)d_ws; size_t off = 0;
  auto F32 = [&](size_t n) -> float* { float* pp = (float*)(wsb + off); off = (off + n * 4 + 255) & ~(size_t)255; return pp; };
  auto BF  = [&](size_t n) -> __bf16* { __bf16* pp = (__bf16*)(wsb + off); off = (off + n * 2 + 255) & ~(size_t)255; return pp; };

  float* part = F32(8192); float* mv = F32(32); float* mvF = F32(32);
  float* a1 = F32(4096); float* d1 = F32(4096);
  float* a2 = F32(4096); float* d2 = F32(4096);
  float* af = F32(4096); float* df = F32(4096);
  float* wq1e = F32(4096); float* wq2e = F32(4096);
  float* cq1 = F32(16); float* cq2 = F32(16);
  float* bv1e = F32(4096); float* bv2e = F32(4096); float* bf1e = F32(8192);
  float* q1 = F32(262144); float* q2 = F32(262144);
  float* s2buf = F32(262144); float* s1buf = F32(262144);
  float* t2 = F32(262144); float* t1 = F32(1048576);
  float* ctx2 = F32(262144); float* ctx1 = F32(1048576);
  float* partF = F32(8192);
  __bf16* wv1b = BF((size_t)16 * 65536); __bf16* wv2b = BF((size_t)16 * 65536);
  __bf16* wout1f = BF(65536); __bf16* wout2f = BF(65536);
  __bf16* wf1b = BF((size_t)16 * 131072); __bf16* wf2f = BF(131072);

  k_stats_part<<<4096, 256, 0, stream>>>(x, part);
  k_stats_final<<<16, 256, 0, stream>>>(part, mv);
  k_prep_affine<<<16, 256, 0, stream>>>(mv, g1, b1, g2, b2, Wqkv1, Wqkv2, a1, d1, a2, d2, wq1e, wq2e);
  k_bias_eff<<<dim3(256, 16), 256, 0, stream>>>(Wqkv1, 256, 257, bqkv1, d1, bv1e);
  k_bias_eff<<<dim3(256, 16), 256, 0, stream>>>(Wqkv2, 256, 257, bqkv2, d2, bv2e);
  k_bias_eff<<<dim3(1, 16), 256, 0, stream>>>(Wqkv1, 256, 0, bqkv1, d1, cq1);
  k_bias_eff<<<dim3(1, 16), 256, 0, stream>>>(Wqkv2, 256, 0, bqkv2, d2, cq2);
  k_build_frags<<<4096, 256, 0, stream>>>(Wqkv1 + 257 * 256, 256, 256, a1, wv1b);
  k_build_frags<<<4096, 256, 0, stream>>>(Wqkv2 + 257 * 256, 256, 256, a2, wv2b);
  k_build_frags<<<256, 256, 0, stream>>>(Wout1, 256, 256, nullptr, wout1f);
  k_build_frags<<<256, 256, 0, stream>>>(Wout2, 256, 256, nullptr, wout2f);
  k_build_frags<<<512, 256, 0, stream>>>(Wf2, 256, 512, nullptr, wf2f);
  k_q<<<1024, 256, 0, stream>>>(x, wq1e, wq2e, cq1, cq2, q1, q2);
  k_softmax2<<<1024, 256, 0, stream>>>(q2, s2buf);
  k_softmax1<<<4096, 64, 0, stream>>>(q1, s1buf);
  k_traw<<<4096, 256, 0, stream>>>(x, s1buf, s2buf, a1, d1, a2, d2, t1, t2);
  k_ctx2<<<4096, 64, 0, stream>>>(Wqkv2, bqkv2, t2, ctx2);
  k_ctx1<<<4096, 256, 0, stream>>>(Wqkv1, bqkv1, t1, ctx1);
  k_main<<<4096, 256, 0, stream>>>(x, wv1b, wv2b, wout1f, wout2f, bv1e, bv2e,
                                   bout1, bout2, ctx1, ctx2, xout, partF);
  k_stats_final<<<16, 256, 0, stream>>>(partF, mvF);
  k_prep_affineF<<<16, 256, 0, stream>>>(mvF, gf, bfv, af, df);
  k_bias_eff<<<dim3(512, 16), 256, 0, stream>>>(Wf1, 256, 0, bf1, df, bf1e);
  k_build_frags<<<8192, 256, 0, stream>>>(Wf1, 512, 256, af, wf1b);
  k_ffn<<<8192, 256, 0, stream>>>(xout, wf1b, wf2f, bf1e, bf2);
}
